// AttentionHead_32315333935830
// MI455X (gfx1250) — matrix operand order repaired and checked by reading, not compiled, not run on hardware
//
#include <hip/hip_runtime.h>

typedef __attribute__((ext_vector_type(16))) __bf16 v16bf;
typedef __attribute__((ext_vector_type(8)))  float  v8f;
typedef __attribute__((ext_vector_type(8)))  unsigned int v8u;
typedef __attribute__((ext_vector_type(4)))  unsigned int v4u;
typedef __attribute__((ext_vector_type(4)))  int v4i;
typedef __attribute__((ext_vector_type(8)))  int v8i;

union frag16 { v16bf bf; v8u u; };

#if defined(__gfx1250__) && __has_builtin(__builtin_amdgcn_tensor_load_to_lds)
#define USE_TDM 1
#else
#define USE_TDM 0
#endif

__device__ __forceinline__ unsigned short f2bf(float f) {
    unsigned int u = __float_as_uint(f);
    u = (u + 0x7FFFu + ((u >> 16) & 1u)) >> 16;   // round-to-nearest-even
    return (unsigned short)u;
}

__device__ __forceinline__ v8f wmma_bf16(const frag16& a, const frag16& b, v8f c) {
    return __builtin_amdgcn_wmma_f32_16x16x32_bf16(false, a.bf, false, b.bf,
                                                   (short)0, c, false, false);
}

__device__ __forceinline__ unsigned lds_off(const void* p) {
    return (unsigned)(unsigned long long)(uintptr_t)p;   // low 32 bits = LDS byte offset
}

#if USE_TDM
// TDM 2D tile load: global (row-major, row_stride elems of 2B) -> LDS,
// with LDS row padding (pad codes per D# group1 encoding).
__device__ __forceinline__ void tdm_load_2d(
    unsigned lds_addr, const void* gaddr,
    unsigned tile_w, unsigned tile_h,          // elements (2B each)
    unsigned tensor_w, unsigned tensor_h,      // elements
    unsigned row_stride,                       // elements
    unsigned pad_interval_code, unsigned pad_amount_code)
{
    unsigned long long ga = (unsigned long long)(uintptr_t)gaddr;
    v4u g0;
    g0[0] = 1u;                                            // count=1, user mode
    g0[1] = lds_addr;                                      // lds_addr [63:32]
    g0[2] = (unsigned)(ga & 0xFFFFFFFFu);                  // global_addr lo
    g0[3] = (unsigned)((ga >> 32) & 0x1FFFFFFu) | (2u << 30); // addr hi + type=2
    v8i g1;
    g1[0] = (int)((1u << 16) |                             // data_size=1 (2 bytes)
                  (1u << 20) |                             // pad_enable
                  (pad_interval_code << 22) |
                  (pad_amount_code << 25));
    g1[1] = (int)((tensor_w & 0xFFFFu) << 16);             // tensor_dim0[15:0]
    g1[2] = (int)(((tensor_w >> 16) & 0xFFFFu) |
                  ((tensor_h & 0xFFFFu) << 16));           // dim0 hi | dim1 lo
    g1[3] = (int)(((tensor_h >> 16) & 0xFFFFu) |
                  ((tile_w & 0xFFFFu) << 16));             // dim1 hi | tile_dim0
    g1[4] = (int)(tile_h & 0xFFFFu);                       // tile_dim1 (tile_dim2=0)
    g1[5] = (int)row_stride;                               // tensor_dim0_stride lo
    g1[6] = 0;                                             // stride hi, dim1_stride
    g1[7] = 0;
    v4i g2 = {0, 0, 0, 0};
    v4i g3 = {0, 0, 0, 0};
    v8i g4 = {0, 0, 0, 0, 0, 0, 0, 0};                     // unused group (6-arg form)
    __builtin_amdgcn_tensor_load_to_lds(g0, g1, g2, g3, g4, 0);
}
#endif

// A fragment: 16x32 bf16 from LDS stored [row][k], row-major.
__device__ __forceinline__ void load_a_frag(const unsigned short* base, int stride,
                                            int row0, int k0, frag16& f) {
    int lane = threadIdx.x & 31;
    int half = lane >> 4;
    int r    = lane & 15;
    const unsigned short* p = base + (size_t)(row0 + r) * stride + k0 + (half ? 16 : 0);
#pragma unroll
    for (int i = 0; i < 8; ++i) {
        f.u[i]     = *(const unsigned int*)(p + 2 * i);
    }
}

// B fragment: 32x16 bf16, B column n stored as LDS row n ([n][k] layout).
__device__ __forceinline__ void load_b_frag(const unsigned short* base, int stride,
                                            int col0, int k0, frag16& f) {
    int lane = threadIdx.x & 31;
    int half = lane >> 4;
    int c    = lane & 15;
    const unsigned short* p = base + (size_t)(col0 + c) * stride + k0 + (half ? 16 : 0);
#pragma unroll
    for (int j = 0; j < 8; ++j)
        f.u[j] = *(const unsigned int*)(p + 2 * j);
}

// B fragment from LDS stored [k][n] (V tile: rows = key, cols = embed)
__device__ __forceinline__ void load_b_frag_t(const unsigned short* base, int stride,
                                              int n0, int k0, frag16& f) {
    int lane = threadIdx.x & 31;
    int half = lane >> 4;
    int c    = lane & 15;
    const unsigned short* p = base + (size_t)(k0 + (half ? 16 : 0)) * stride + n0 + c;
#pragma unroll
    for (int j = 0; j < 8; ++j) {
        unsigned int lo = p[(2 * j) * stride];
        unsigned int hi = p[(2 * j + 1) * stride];
        f.u[j] = lo | (hi << 16);
    }
}

#define DIMD  1024
#define NROWS 8192   // 4 * 2048

// ---------------- Kernel 1: fused QKV projection ----------------
__global__ __launch_bounds__(256) void qkv_proj_kernel(
    const float* __restrict__ x,
    const float* __restrict__ Wq, const float* __restrict__ bq,
    const float* __restrict__ Wk, const float* __restrict__ bk,
    const float* __restrict__ Wv, const float* __restrict__ bv,
    unsigned short* __restrict__ Q, unsigned short* __restrict__ K,
    unsigned short* __restrict__ V)
{
    const int which = blockIdx.z;
    const float* W    = (which == 0) ? Wq : (which == 1) ? Wk : Wv;
    const float* bias = (which == 0) ? bq : (which == 1) ? bk : bv;
    unsigned short* out = (which == 0) ? Q : (which == 1) ? K : V;

    const int mbase = blockIdx.y * 128;
    const int nbase = blockIdx.x * 64;

    __shared__ unsigned short Xs[128][36];
    __shared__ unsigned short Ws[64][36];

    const int tid  = threadIdx.x;
    const int w    = tid >> 5;
    const int lane = tid & 31;
    const int wm   = w & 3;
    const int wn   = w >> 2;
    const int half = lane >> 4;
    const int lc   = lane & 15;

    v8f acc[2][2] = {};

    for (int k0 = 0; k0 < DIMD; k0 += 32) {
        for (int t = tid; t < 128 * 16; t += 256) {
            int rr = t >> 4, cc = (t & 15) * 2;
            float2 xv = *(const float2*)(x + (size_t)(mbase + rr) * DIMD + k0 + cc);
            *(unsigned int*)&Xs[rr][cc] =
                (unsigned int)f2bf(xv.x) | ((unsigned int)f2bf(xv.y) << 16);
        }
        for (int t = tid; t < 64 * 16; t += 256) {
            int rr = t >> 4, cc = (t & 15) * 2;
            float2 wv = *(const float2*)(W + (size_t)(nbase + rr) * DIMD + k0 + cc);
            *(unsigned int*)&Ws[rr][cc] =
                (unsigned int)f2bf(wv.x) | ((unsigned int)f2bf(wv.y) << 16);
        }
        __syncthreads();

        frag16 a0, a1, b0, b1;
        load_a_frag(&Xs[0][0], 36, 32 * wm,      0, a0);
        load_a_frag(&Xs[0][0], 36, 32 * wm + 16, 0, a1);
        load_b_frag(&Ws[0][0], 36, 32 * wn,      0, b0);
        load_b_frag(&Ws[0][0], 36, 32 * wn + 16, 0, b1);
        acc[0][0] = wmma_bf16(a0, b0, acc[0][0]);
        acc[0][1] = wmma_bf16(a0, b1, acc[0][1]);
        acc[1][0] = wmma_bf16(a1, b0, acc[1][0]);
        acc[1][1] = wmma_bf16(a1, b1, acc[1][1]);
        __syncthreads();
    }

#pragma unroll
    for (int mt = 0; mt < 2; ++mt)
#pragma unroll
        for (int nt = 0; nt < 2; ++nt) {
            int col = nbase + 32 * wn + 16 * nt + lc;
            float bb = bias[col];
#pragma unroll
            for (int i = 0; i < 8; ++i) {
                int row = mbase + 32 * wm + 16 * mt + i + 8 * half;
                out[(size_t)row * DIMD + col] = f2bf(acc[mt][nt][i] + bb);
            }
        }
}

// ---------------- Kernel 2: flash attention ----------------
#define SLEN 2048
#define MQ   32
#define NKT  64
#define DSL  64
#define ESL  128

__global__ __launch_bounds__(256) void flash_attn_kernel(
    const unsigned short* __restrict__ Q, const unsigned short* __restrict__ K,
    const unsigned short* __restrict__ V, float* __restrict__ out)
{
    const int b     = blockIdx.y;
    const int qbase = blockIdx.x * MQ;
    const unsigned short* Qb = Q + (size_t)b * SLEN * DIMD;
    const unsigned short* Kb = K + (size_t)b * SLEN * DIMD;
    const unsigned short* Vb = V + (size_t)b * SLEN * DIMD;
    float* outb = out + (size_t)b * SLEN * DIMD;

    __shared__ unsigned short Qs[MQ][DSL + 4];   // rows padded: 32 DW + 2 DW
    __shared__ unsigned short Ks[NKT][DSL + 4];
    __shared__ unsigned short Ps[MQ][NKT + 4];
    __shared__ unsigned short Vs[NKT][ESL + 4];  // rows padded: 64 DW + 2 DW
    __shared__ float redmax[MQ][4];
    __shared__ float redsum[MQ][4];
    __shared__ float mstate[MQ];
    __shared__ float lstate[MQ];

    const int tid  = threadIdx.x;
    const int w    = tid >> 5;
    const int lane = tid & 31;
    const int wm   = w & 1;
    const int wn   = w >> 1;
    const int half = lane >> 4;
    const int lc   = lane & 15;

    if (tid < MQ) { mstate[tid] = -1e30f; lstate[tid] = 0.0f; }
    __syncthreads();

    v8f o[8][2] = {};

    for (int j = 0; j < SLEN; j += NKT) {
        // ---- S = (Q Kt) * scale ----
        v8f s = {};
        for (int d0 = 0; d0 < DIMD; d0 += DSL) {
#if USE_TDM
            // pad codes: interval 4 -> 32 DWORDs (64 bf16), amount 1 -> 2 DWORDs (4 bf16)
            if (w == 0)
                tdm_load_2d(lds_off(&Qs[0][0]), Qb + (size_t)qbase * DIMD + d0,
                            DSL, MQ, DIMD, SLEN, DIMD, 4, 1);
            if (w == 1)
                tdm_load_2d(lds_off(&Ks[0][0]), Kb + (size_t)j * DIMD + d0,
                            DSL, NKT, DIMD, SLEN, DIMD, 4, 1);
            __builtin_amdgcn_s_wait_tensorcnt(0);
#else
            for (int t = tid; t < MQ * (DSL / 2); t += 256) {
                int rr = t >> 5, cc = (t & 31) * 2;
                *(unsigned int*)&Qs[rr][cc] =
                    *(const unsigned int*)(Qb + (size_t)(qbase + rr) * DIMD + d0 + cc);
            }
            for (int t = tid; t < NKT * (DSL / 2); t += 256) {
                int rr = t >> 5, cc = (t & 31) * 2;
                *(unsigned int*)&Ks[rr][cc] =
                    *(const unsigned int*)(Kb + (size_t)(j + rr) * DIMD + d0 + cc);
            }
#endif
            __syncthreads();
            frag16 qa, kb;
#pragma unroll
            for (int kk = 0; kk < DSL; kk += 32) {
                load_a_frag(&Qs[0][0], DSL + 4, 16 * wm, kk, qa);
                load_b_frag(&Ks[0][0], DSL + 4, 16 * wn, kk, kb);
                s = wmma_bf16(qa, kb, s);
            }
            __syncthreads();
        }
#pragma unroll
        for (int i = 0; i < 8; ++i) s[i] *= 0.03125f;  // 1/sqrt(1024)

        // ---- per-wave row max ----
#pragma unroll
        for (int i = 0; i < 8; ++i) {
            float v = s[i];
            v = fmaxf(v, __shfl_xor(v, 1));
            v = fmaxf(v, __shfl_xor(v, 2));
            v = fmaxf(v, __shfl_xor(v, 4));
            v = fmaxf(v, __shfl_xor(v, 8));
            if (lc == 0) redmax[16 * wm + i + 8 * half][wn] = v;
        }
        __syncthreads();

        // ---- online softmax ----
        float alpha[8], mnewa[8];
#pragma unroll
        for (int i = 0; i < 8; ++i) {
            int row = 16 * wm + i + 8 * half;
            float tmax = fmaxf(fmaxf(redmax[row][0], redmax[row][1]),
                               fmaxf(redmax[row][2], redmax[row][3]));
            float mold = mstate[row];
            float mnew = fmaxf(mold, tmax);
            mnewa[i]   = mnew;
            alpha[i]   = __expf(mold - mnew);
            float p    = __expf(s[i] - mnew);
            s[i] = p;
            float sum = p;
            sum += __shfl_xor(sum, 1);
            sum += __shfl_xor(sum, 2);
            sum += __shfl_xor(sum, 4);
            sum += __shfl_xor(sum, 8);
            if (lc == 0) redsum[row][wn] = sum;
        }
        __syncthreads();

        if (wn == 0 && lc == 0) {
#pragma unroll
            for (int i = 0; i < 8; ++i) {
                int row = 16 * wm + i + 8 * half;
                float tot = redsum[row][0] + redsum[row][1] +
                            redsum[row][2] + redsum[row][3];
                lstate[row] = lstate[row] * alpha[i] + tot;
                mstate[row] = mnewa[i];
            }
        }
#pragma unroll
        for (int i = 0; i < 8; ++i)
            Ps[16 * wm + i + 8 * half][16 * wn + lc] = f2bf(s[i]);
#pragma unroll
        for (int si = 0; si < 8; ++si)
#pragma unroll
            for (int ct = 0; ct < 2; ++ct)
#pragma unroll
                for (int i = 0; i < 8; ++i) o[si][ct][i] *= alpha[i];
        __syncthreads();

        // ---- O += P @ V ----
        frag16 pa0, pa1;
        load_a_frag(&Ps[0][0], NKT + 4, 16 * wm, 0,  pa0);
        load_a_frag(&Ps[0][0], NKT + 4, 16 * wm, 32, pa1);
        for (int si = 0; si < 8; ++si) {
            int e0 = si * ESL;
#if USE_TDM
            // pad codes: interval 5 -> 64 DWORDs (128 bf16), amount 1 -> 2 DWORDs
            if (w == 0)
                tdm_load_2d(lds_off(&Vs[0][0]), Vb + (size_t)j * DIMD + e0,
                            ESL, NKT, DIMD, SLEN, DIMD, 5, 1);
            __builtin_amdgcn_s_wait_tensorcnt(0);
#else
            for (int t = tid; t < NKT * (ESL / 2); t += 256) {
                int rr = t >> 6, cc = (t & 63) * 2;
                *(unsigned int*)&Vs[rr][cc] =
                    *(const unsigned int*)(Vb + (size_t)(j + rr) * DIMD + e0 + cc);
            }
#endif
            __syncthreads();
#pragma unroll
            for (int ct = 0; ct < 2; ++ct) {
                int cb = 32 * wn + 16 * ct;
                frag16 vb;
                load_b_frag_t(&Vs[0][0], ESL + 4, cb, 0, vb);
                o[si][ct] = wmma_bf16(pa0, vb, o[si][ct]);
                load_b_frag_t(&Vs[0][0], ESL + 4, cb, 32, vb);
                o[si][ct] = wmma_bf16(pa1, vb, o[si][ct]);
            }
            __syncthreads();
        }
    }

    // ---- finalize ----
    float linv[8];
#pragma unroll
    for (int i = 0; i < 8; ++i)
        linv[i] = 1.0f / lstate[16 * wm + i + 8 * half];
#pragma unroll
    for (int si = 0; si < 8; ++si)
#pragma unroll
        for (int ct = 0; ct < 2; ++ct) {
            int col = si * ESL + 32 * wn + 16 * ct + lc;
#pragma unroll
            for (int i = 0; i < 8; ++i) {
                int row = qbase + 16 * wm + i + 8 * half;
                outb[(size_t)row * DIMD + col] = o[si][ct][i] * linv[i];
            }
        }
}

extern "C" void kernel_launch(void* const* d_in, const int* in_sizes, int n_in,
                              void* d_out, int out_size, void* d_ws, size_t ws_size,
                              hipStream_t stream) {
    (void)in_sizes; (void)n_in; (void)out_size; (void)ws_size;
    const float* x  = (const float*)d_in[0];
    const float* Wq = (const float*)d_in[1];
    const float* bq = (const float*)d_in[2];
    const float* Wk = (const float*)d_in[3];
    const float* bk = (const float*)d_in[4];
    const float* Wv = (const float*)d_in[5];
    const float* bv = (const float*)d_in[6];

    unsigned short* Qw = (unsigned short*)d_ws;
    unsigned short* Kw = Qw + (size_t)NROWS * DIMD;
    unsigned short* Vw = Kw + (size_t)NROWS * DIMD;

    qkv_proj_kernel<<<dim3(DIMD / 64, NROWS / 128, 3), 256, 0, stream>>>(
        x, Wq, bq, Wk, bk, Wv, bv, Qw, Kw, Vw);

    flash_attn_kernel<<<dim3(SLEN / MQ, 4), 256, 0, stream>>>(
        Qw, Kw, Vw, (float*)d_out);
}
